// Seq2seq_attn_67954972557714
// MI455X (gfx1250) — matrix operand order repaired and checked by reading, not compiled, not run on hardware
//
#include <hip/hip_runtime.h>
#include <hip/hip_bf16.h>
#include <stdint.h>

#define H      256
#define L      2
#define NB     32
#define SLEN   50
#define TLEN   50
#define VOC    32000
#define G3H    768
#define NTHR   256   // 8 wave32s

typedef __attribute__((ext_vector_type(16))) __bf16         v16bf;
typedef __attribute__((ext_vector_type(16))) unsigned short v16us;
typedef __attribute__((ext_vector_type(8)))  float          v8f;

// ---------- helpers ----------
__device__ __forceinline__ unsigned short f32_to_bf16(float f) {
  unsigned u = __float_as_uint(f);
  unsigned r = (u >> 16) & 1u;
  u += 0x7FFFu + r;              // round-to-nearest-even
  return (unsigned short)(u >> 16);
}

__device__ __forceinline__ unsigned ordf(float f) {
  unsigned u = __float_as_uint(f);
  return (u & 0x80000000u) ? ~u : (u | 0x80000000u);  // order-preserving map
}

// A fragment: 16x32 bf16, row-major source with stride lda (elements).
// lanes 0-15: row M=lane, elems 0..15 -> K=0..15
// lanes 16-31: row M=lane-16, elems 0..15 -> K=16..31
__device__ __forceinline__ v16bf load_A_bf16(const unsigned short* base, int lda, int lane) {
  int half = (lane >> 4) & 1;
  int m    = lane & 15;
  const unsigned short* row = base + m * lda;
  int k0 = half * 16;
  int k1 = half * 16 + 8;
  v16us v;
#pragma unroll
  for (int e = 0; e < 8; ++e) { v[e] = row[k0 + e]; v[8 + e] = row[k1 + e]; }
  return __builtin_bit_cast(v16bf, v);
}

// B fragment: 32x16 bf16 where B[k][n] = W[n0+n][k0+k], W row-major (out,in), stride ldw.
// lanes 0-15: N=lane, elems = K=k0..k0+15 ; lanes 16-31: N=lane-16, K=k0+16..k0+31
__device__ __forceinline__ v16bf load_B_bf16(const unsigned short* w, int ldw,
                                             int n0, int k0, int lane) {
  int half = (lane >> 4) & 1;
  int n    = lane & 15;
  const unsigned short* row = w + (size_t)(n0 + n) * ldw + k0 + half * 16;
  v16us v;
#pragma unroll
  for (int e = 0; e < 16; ++e) v[e] = row[e];
  return __builtin_bit_cast(v16bf, v);
}

// C = A(32..xK bf16, in LDS) @ W^T(bf16, global) + bias ; C f32 (Mtiles*16 x Ntiles*16), stride ldc.
// Assumes blockDim.x == 256 (8 waves). Tile predicate is wave-uniform -> EXEC all ones at WMMA.
__device__ __forceinline__ void mm_bf16(const unsigned short* __restrict__ A, int lda,
                                        const unsigned short* __restrict__ W, int ldw,
                                        const float* __restrict__ bias,
                                        float* __restrict__ C, int ldc,
                                        int Mtiles, int Ntiles, int K, int tid) {
  int wave = tid >> 5, lane = tid & 31;
  int half = lane >> 4, nn = lane & 15;
  int nT = Mtiles * Ntiles;
  for (int tile = wave; tile < nT; tile += 8) {
    int rt = tile % Mtiles;
    int ct = tile / Mtiles;
    v8f acc = {};
    for (int kk = 0; kk < K; kk += 32) {
      v16bf a = load_A_bf16(A + rt * 16 * lda + kk, lda, lane);
      v16bf b = load_B_bf16(W, ldw, ct * 16, kk, lane);
      acc = __builtin_amdgcn_wmma_f32_16x16x32_bf16(false, a, false, b,
                                                    (short)0, acc, false, false);
    }
    float bv = bias ? bias[ct * 16 + nn] : 0.0f;
#pragma unroll
    for (int r = 0; r < 8; ++r)
      C[(size_t)(rt * 16 + half * 8 + r) * ldc + (ct * 16 + nn)] = acc[r] + bv;
  }
}

// GRU gate math (PyTorch convention). gi/gh: (NB x 3H) in LDS. h: global f32 for this layer.
// Also writes new h as bf16 into xbf (input of next layer) and optionally top output.
__device__ __forceinline__ void gru_gates(const float* __restrict__ gi,
                                          const float* __restrict__ gh,
                                          float* __restrict__ hl,
                                          unsigned short* __restrict__ xbf,
                                          float* __restrict__ top_out, int top_stride,
                                          int tid) {
  for (int i = tid; i < NB * H; i += NTHR) {
    int n = i >> 8, j = i & 255;
    const float* gin = gi + n * G3H;
    const float* ghn = gh + n * G3H;
    float r  = 1.0f / (1.0f + __expf(-(gin[j]         + ghn[j])));
    float z  = 1.0f / (1.0f + __expf(-(gin[j + H]     + ghn[j + H])));
    float nc = tanhf(gin[j + 2 * H] + r * ghn[j + 2 * H]);
    float hnew = (1.0f - z) * nc + z * hl[i];
    hl[i]  = hnew;
    xbf[i] = f32_to_bf16(hnew);
    if (top_out) top_out[(size_t)n * top_stride + j] = hnew;
  }
}

// ---------- kernels ----------
__global__ void cvt_bf16_kernel(const float* __restrict__ src,
                                unsigned short* __restrict__ dst, int n) {
  int i = blockIdx.x * blockDim.x + threadIdx.x;
  int stride = gridDim.x * blockDim.x;
  for (; i < n; i += stride) dst[i] = f32_to_bf16(src[i]);
}

// Full encoder: one workgroup, 50 sequential steps, 2 layers, WMMA GRU.
__global__ void encoder_kernel(const int* __restrict__ xids,
                               const float* __restrict__ fr_emb,
                               const unsigned short* __restrict__ WihBf,
                               const unsigned short* __restrict__ WhhBf,
                               const float* __restrict__ bih,
                               const float* __restrict__ bhh,
                               float* __restrict__ enc_outs,
                               float* __restrict__ h,
                               unsigned long long* __restrict__ tok_pack) {
  extern __shared__ char smem[];
  unsigned short* xbf = (unsigned short*)smem;             // 16 KB
  unsigned short* hbf = (unsigned short*)(smem + 16384);   // 16 KB
  float* gi = (float*)(smem + 32768);                      // 96 KB
  float* gh = (float*)(smem + 131072);                     // 96 KB
  int tid = threadIdx.x;

  for (int i = tid; i < L * NB * H; i += NTHR) h[i] = 0.0f;
  if (tid < NB) tok_pack[tid] = 0x00000000FFFFFFFFull;     // value=-inf(ordered 0), token 0
  __syncthreads();

  for (int t = 0; t < SLEN; ++t) {
    for (int i = tid; i < NB * H; i += NTHR) {
      int n = i >> 8, j = i & 255;
      int id = xids[n * SLEN + t];
      xbf[i] = f32_to_bf16(fr_emb[(size_t)id * H + j]);
    }
    __syncthreads();
    for (int l = 0; l < L; ++l) {
      for (int i = tid; i < NB * H; i += NTHR) hbf[i] = f32_to_bf16(h[l * NB * H + i]);
      __syncthreads();
      mm_bf16(xbf, H, WihBf + (size_t)l * G3H * H, H, bih + l * G3H, gi, G3H, 2, 48, H, tid);
      mm_bf16(hbf, H, WhhBf + (size_t)l * G3H * H, H, bhh + l * G3H, gh, G3H, 2, 48, H, tid);
      __syncthreads();
      gru_gates(gi, gh, h + l * NB * H, xbf,
                (l == L - 1) ? (enc_outs + (size_t)t * H) : nullptr, SLEN * H, tid);
      __syncthreads();
    }
  }
}

// One decoder step: embed greedy token, 2-layer GRU, dot attention, fc1. One workgroup.
__global__ void dec_step_kernel(int t,
                                const float* __restrict__ en_emb,
                                const unsigned short* __restrict__ WihBf,
                                const unsigned short* __restrict__ WhhBf,
                                const float* __restrict__ bih,
                                const float* __restrict__ bhh,
                                const unsigned short* __restrict__ WqBf,
                                const float* __restrict__ bq,
                                const unsigned short* __restrict__ WcBf,
                                const float* __restrict__ bc,
                                const float* __restrict__ enc_outs,
                                float* __restrict__ h,
                                unsigned short* __restrict__ fc1bf,
                                unsigned long long* __restrict__ tok_pack,
                                float* __restrict__ attn_out) {
  extern __shared__ char smem[];
  unsigned short* xbf = (unsigned short*)smem;
  unsigned short* hbf = (unsigned short*)(smem + 16384);
  float* gi = (float*)(smem + 32768);
  float* gh = (float*)(smem + 131072);
  // post-GRU overlays (gi/gh regions are dead by then)
  unsigned short* rnnbf = (unsigned short*)(smem + 32768);   // 16 KB
  float* qf   = (float*)(smem + 49152);                      // 32 KB
  float* simf = (float*)(smem + 81920);                      // 6.25 KB
  float* wf   = (float*)(smem + 88320);                      // 6.25 KB
  float* ctxf = (float*)(smem + 94720);                      // 32 KB
  unsigned short* catbf = (unsigned short*)(smem + 127488);  // 32 KB
  float* fc1f = (float*)(smem + 160256);                     // 32 KB
  __shared__ int tok_s[NB];
  int tid = threadIdx.x;

  if (tid < NB)
    tok_s[tid] = (int)(0xFFFFFFFFu - (unsigned)(tok_pack[tid] & 0xFFFFFFFFull));
  __syncthreads();
  if (tid < NB) tok_pack[tid] = 0x00000000FFFFFFFFull;  // reset for this step's fc2

  for (int i = tid; i < NB * H; i += NTHR) {
    int n = i >> 8, j = i & 255;
    xbf[i] = f32_to_bf16(en_emb[(size_t)tok_s[n] * H + j]);
  }
  __syncthreads();

  for (int l = 0; l < L; ++l) {
    for (int i = tid; i < NB * H; i += NTHR) hbf[i] = f32_to_bf16(h[l * NB * H + i]);
    __syncthreads();
    mm_bf16(xbf, H, WihBf + (size_t)l * G3H * H, H, bih + l * G3H, gi, G3H, 2, 48, H, tid);
    mm_bf16(hbf, H, WhhBf + (size_t)l * G3H * H, H, bhh + l * G3H, gh, G3H, 2, 48, H, tid);
    __syncthreads();
    gru_gates(gi, gh, h + l * NB * H, xbf, nullptr, 0, tid);
    __syncthreads();
  }

  const float* rnn = h + (L - 1) * NB * H;   // top-layer output
  for (int i = tid; i < NB * H; i += NTHR) rnnbf[i] = f32_to_bf16(rnn[i]);
  __syncthreads();

  // q = rnn @ Wq^T + bq
  mm_bf16(rnnbf, H, WqBf, H, bq, qf, H, 2, 16, H, tid);
  __syncthreads();

  // sim[n][s] = <enc_outs[n,s,:], q[n,:]>
  for (int p = tid; p < NB * SLEN; p += NTHR) {
    int n = p / SLEN, s = p % SLEN;
    const float* e  = enc_outs + (size_t)(n * SLEN + s) * H;
    const float* qn = qf + n * H;
    float acc = 0.0f;
    for (int j = 0; j < H; ++j) acc += e[j] * qn[j];
    simf[p] = acc;
  }
  __syncthreads();

  if (tid < NB) {                 // softmax over S per batch row
    float mx = -3.0e38f;
    for (int s = 0; s < SLEN; ++s) mx = fmaxf(mx, simf[tid * SLEN + s]);
    float sum = 0.0f;
    for (int s = 0; s < SLEN; ++s) {
      float e = __expf(simf[tid * SLEN + s] - mx);
      wf[tid * SLEN + s] = e;
      sum += e;
    }
    float inv = 1.0f / sum;
    for (int s = 0; s < SLEN; ++s) wf[tid * SLEN + s] *= inv;
  }
  __syncthreads();

  if (t == TLEN - 1)              // reference keeps only the last step's weights
    for (int p = tid; p < NB * SLEN; p += NTHR) attn_out[p] = wf[p];

  // ctx[n][j] = sum_s w[n][s] * enc_outs[n][s][j]
  for (int i = tid; i < NB * H; i += NTHR) {
    int n = i >> 8, j = i & 255;
    float a = 0.0f;
    for (int s = 0; s < SLEN; ++s)
      a += wf[n * SLEN + s] * enc_outs[(size_t)(n * SLEN + s) * H + j];
    ctxf[i] = a;
  }
  __syncthreads();

  // concat(rnn, ctx) -> bf16
  for (int i = tid; i < NB * 2 * H; i += NTHR) {
    int n = i >> 9, j = i & 511;
    float v = (j < H) ? rnn[n * H + j] : ctxf[n * H + (j - H)];
    catbf[i] = f32_to_bf16(v);
  }
  __syncthreads();

  // fc1 = cat @ Wc^T + bc  (K=512)
  mm_bf16(catbf, 2 * H, WcBf, 2 * H, bc, fc1f, H, 2, 16, 2 * H, tid);
  __syncthreads();

  for (int i = tid; i < NB * H; i += NTHR) fc1bf[i] = f32_to_bf16(fc1f[i]);
}

// fc2: logits (32 x 32000) = fc1 @ Wfc^T + bfc, with fused greedy argmax.
// 250 blocks x 8 waves; each wave owns one 16-col tile (x2 row tiles).
__global__ void fc2_kernel(int t,
                           const unsigned short* __restrict__ fc1bf,
                           const unsigned short* __restrict__ WfcBf,
                           const float* __restrict__ bfc,
                           float* __restrict__ logits,
                           unsigned long long* __restrict__ tok_pack) {
  __shared__ unsigned long long best[NB];
  int tid = threadIdx.x;
  if (tid < NB) best[tid] = 0ull;
  __syncthreads();

  int wave = tid >> 5, lane = tid & 31;
  int half = lane >> 4, nn = lane & 15;
  int colbase = blockIdx.x * 128 + wave * 16;

  for (int rt = 0; rt < 2; ++rt) {
    v8f acc = {};
    for (int kk = 0; kk < H; kk += 32) {
      v16bf a = load_A_bf16(fc1bf + rt * 16 * H + kk, H, lane);
      v16bf b = load_B_bf16(WfcBf, H, colbase, kk, lane);
      acc = __builtin_amdgcn_wmma_f32_16x16x32_bf16(false, a, false, b,
                                                    (short)0, acc, false, false);
    }
    int col = colbase + nn;
    float bv = bfc[col];
#pragma unroll
    for (int r = 0; r < 8; ++r) {
      int row = rt * 16 + half * 8 + r;
      float val = acc[r] + bv;
      __builtin_nontemporal_store(val, &logits[((size_t)row * TLEN + t) * VOC + col]);
      unsigned long long pk = ((unsigned long long)ordf(val) << 32)
                            | (unsigned long long)(0xFFFFFFFFu - (unsigned)col);
      atomicMax(&best[row], pk);   // DS u64 max
    }
  }
  __syncthreads();
  if (tid < NB) atomicMax(&tok_pack[tid], best[tid]);
}

// ---------- host ----------
extern "C" void kernel_launch(void* const* d_in, const int* in_sizes, int n_in,
                              void* d_out, int out_size, void* d_ws, size_t ws_size,
                              hipStream_t stream) {
  const int*   x       = (const int*)  d_in[0];
  const float* fr_emb  = (const float*)d_in[1];
  const float* en_emb  = (const float*)d_in[2];
  const float* enc_Wih = (const float*)d_in[3];
  const float* enc_Whh = (const float*)d_in[4];
  const float* enc_bih = (const float*)d_in[5];
  const float* enc_bhh = (const float*)d_in[6];
  const float* dec_Wih = (const float*)d_in[7];
  const float* dec_Whh = (const float*)d_in[8];
  const float* dec_bih = (const float*)d_in[9];
  const float* dec_bhh = (const float*)d_in[10];
  const float* Wq      = (const float*)d_in[11];
  const float* bq      = (const float*)d_in[12];
  const float* Wc      = (const float*)d_in[13];
  const float* bc      = (const float*)d_in[14];
  const float* Wfc     = (const float*)d_in[15];
  const float* bfc     = (const float*)d_in[16];
  float* out = (float*)d_out;

  // workspace layout (bytes)
  char* ws = (char*)d_ws;
  unsigned short* encWihBf = (unsigned short*)(ws + 0);          // 786432 B
  unsigned short* encWhhBf = (unsigned short*)(ws + 786432);
  unsigned short* decWihBf = (unsigned short*)(ws + 1572864);
  unsigned short* decWhhBf = (unsigned short*)(ws + 2359296);
  unsigned short* WqBf     = (unsigned short*)(ws + 3145728);    // 131072 B
  unsigned short* WcBf     = (unsigned short*)(ws + 3276800);    // 262144 B
  unsigned short* WfcBf    = (unsigned short*)(ws + 3538944);    // 16384000 B
  float*          enc_outs = (float*)(ws + 19922944);            // 1638400 B
  float*          hbuf     = (float*)(ws + 21561344);            // 65536 B
  unsigned short* fc1bf    = (unsigned short*)(ws + 21626880);   // 16384 B
  unsigned long long* tokp = (unsigned long long*)(ws + 21643264);

  const size_t HFIN_OFF = (size_t)NB * TLEN * VOC;               // 51,200,000
  const size_t ATTN_OFF = HFIN_OFF + (size_t)L * NB * H;         // 51,216,384
  const int SMEM = 229376;                                       // 224 KB dynamic LDS

  // bf16 weight conversion (Wfc stays L2-resident at 16 MB across all 50 steps)
  cvt_bf16_kernel<<<512, 256, 0, stream>>>(enc_Wih, encWihBf, L * G3H * H);
  cvt_bf16_kernel<<<512, 256, 0, stream>>>(enc_Whh, encWhhBf, L * G3H * H);
  cvt_bf16_kernel<<<512, 256, 0, stream>>>(dec_Wih, decWihBf, L * G3H * H);
  cvt_bf16_kernel<<<512, 256, 0, stream>>>(dec_Whh, decWhhBf, L * G3H * H);
  cvt_bf16_kernel<<<512, 256, 0, stream>>>(Wq, WqBf, H * H);
  cvt_bf16_kernel<<<512, 256, 0, stream>>>(Wc, WcBf, H * 2 * H);
  cvt_bf16_kernel<<<2048, 256, 0, stream>>>(Wfc, WfcBf, VOC * H);

  encoder_kernel<<<1, NTHR, SMEM, stream>>>(x, fr_emb, encWihBf, encWhhBf,
                                            enc_bih, enc_bhh, enc_outs, hbuf, tokp);

  for (int t = 0; t < TLEN; ++t) {
    dec_step_kernel<<<1, NTHR, SMEM, stream>>>(t, en_emb, decWihBf, decWhhBf,
                                               dec_bih, dec_bhh, WqBf, bq, WcBf, bc,
                                               enc_outs, hbuf, fc1bf, tokp,
                                               out + ATTN_OFF);
    fc2_kernel<<<VOC / 128, NTHR, 0, stream>>>(t, fc1bf, WfcBf, bfc, out, tokp);
  }

  // h_fin (L,N,H) -> d_out
  hipMemcpyAsync(out + HFIN_OFF, hbuf, (size_t)L * NB * H * sizeof(float),
                 hipMemcpyDeviceToDevice, stream);
}
